// FancyViT_73461120630931
// MI455X (gfx1250) — matrix operand order repaired and checked by reading, not compiled, not run on hardware
//
#include <hip/hip_runtime.h>
#include <stdint.h>

// ---------------------------------------------------------------------------
// MViTv2/SAM windowed attention with decomposed rel-pos bias, CDNA5 (gfx1250).
// All GEMM stages use v_wmma_f32_16x16x32_bf16 (wave32 WMMA, fp32 accum).
// Pure-copy LDS staging uses global_load_async_to_lds_b128 (ASYNCcnt).
//   B=200 windows, S=14 (196 tokens), C=768, 12 heads, hd=64, scale=1/8.
// Workspace (bf16 = uint16_t), ~241 MB:
//   qb[2400][196][64], kTb[2400][64][196], vb[2400][196][64], ctx[39200][768]
// ---------------------------------------------------------------------------

typedef __attribute__((ext_vector_type(16))) __bf16 v16bf;
typedef __attribute__((ext_vector_type(8)))  float  v8f;

union FragBF { uint32_t u[8]; v16bf v; };

__device__ __forceinline__ uint16_t f2bf(float f) {
  uint32_t u = __float_as_uint(f);
  u += 0x7fffu + ((u >> 16) & 1u);           // round-to-nearest-even
  return (uint16_t)(u >> 16);
}
__device__ __forceinline__ float bf2f(uint32_t h) {
  return __uint_as_float(h << 16);
}
__device__ __forceinline__ uint32_t pack2(float lo, float hi) {
  return (uint32_t)f2bf(lo) | ((uint32_t)f2bf(hi) << 16);
}
// A-fragment pair index (ISA 7.12.2, 16-bit A 16x32): VGPR v holds K pair
// {2v,2v+1} (v<4) or {16+2(v-4),...} (v>=4), +8 for lane half 1.
__device__ __forceinline__ int apair(int v, int half) {
  return (v < 4 ? v : v + 4) + 4 * half;
}
__device__ __forceinline__ v8f wmma_bf16(const FragBF& a, const FragBF& b, v8f c) {
  return __builtin_amdgcn_wmma_f32_16x16x32_bf16(false, a.v, false, b.v,
                                                 (short)0, c, false, false);
}

// Async global->LDS copy, 16 bytes per active lane.  The low 32 bits of a
// generic pointer to __shared__ are the wave-relative LDS byte address
// (ISA 10.2: LDS aperture keeps the offset in addr[31:0]), which is exactly
// the VDST operand of GLOBAL_LOAD_ASYNC_TO_LDS_B128 (15.18.3 op 98).
__device__ __forceinline__ void async_g2l_b128(void* lds, const void* gptr) {
  asm volatile("global_load_async_to_lds_b128 %0, %1, off"
               :: "v"((uint32_t)(uintptr_t)lds), "v"(gptr)
               : "memory");
}
__device__ __forceinline__ void wait_async0() {
  asm volatile("s_wait_asynccnt 0" ::: "memory");
}

#define M_TOK   39200        // 200 * 196
#define C_DIM   768
#define NQKV    2304
#define NHEAD   12
#define HD      64
#define TOK     196
#define TOKP    224          // 196 padded to 14 wmma tiles

// ===========================================================================
// Kernel 1: QKV GEMM.  X(39200x768) fp32 @ Wqkv(768x2304) fp32 + bias.
// 128x128 tile / workgroup, BK=32, 256 threads = 8 waves (4x2 wave grid,
// each wave 32x64 = 2x4 wmma tiles).  Epilogue scatters bf16 q / kT / v.
// ===========================================================================
__global__ __launch_bounds__(256) void fv_qkv_gemm(
    const float* __restrict__ X, const float* __restrict__ W,
    const float* __restrict__ bias,
    uint16_t* __restrict__ qb, uint16_t* __restrict__ kTb,
    uint16_t* __restrict__ vb) {
  __shared__ uint32_t alds[128 * 16];   // [row][kpair] packed bf16 pairs
  __shared__ uint32_t blds[16 * 128];   // [kpair][col]

  const int tid = threadIdx.x, lane = tid & 31, wave = tid >> 5;
  const int bm = blockIdx.x, bn = blockIdx.y;
  const int wr = wave >> 1, wc = wave & 1;
  const int m16 = lane & 15, half = lane >> 4;

  v8f zero = {};
  v8f acc[2][4];
#pragma unroll
  for (int mt = 0; mt < 2; ++mt)
#pragma unroll
    for (int nt = 0; nt < 4; ++nt) acc[mt][nt] = zero;

  for (int k0 = 0; k0 < C_DIM; k0 += 32) {
#pragma unroll
    for (int idx = tid; idx < 128 * 16; idx += 256) {
      int row = idx >> 4, pi = idx & 15;
      int m = bm * 128 + row, k = k0 + 2 * pi;
      float a0 = 0.f, a1 = 0.f;
      if (m < M_TOK) {
        a0 = X[(size_t)m * C_DIM + k];
        a1 = X[(size_t)m * C_DIM + k + 1];
      }
      alds[idx] = pack2(a0, a1);
    }
#pragma unroll
    for (int idx = tid; idx < 16 * 128; idx += 256) {
      int p = idx >> 7, n = idx & 127;
      int k = k0 + 2 * p, ng = bn * 128 + n;
      blds[idx] = pack2(W[(size_t)k * NQKV + ng], W[(size_t)(k + 1) * NQKV + ng]);
    }
    __syncthreads();

    FragBF af[2], bfr[4];
#pragma unroll
    for (int mt = 0; mt < 2; ++mt) {
      int row = wr * 32 + mt * 16 + m16;
#pragma unroll
      for (int v = 0; v < 8; ++v) af[mt].u[v] = alds[row * 16 + apair(v, half)];
    }
#pragma unroll
    for (int nt = 0; nt < 4; ++nt) {
      int n = wc * 64 + nt * 16 + m16;
#pragma unroll
      for (int v = 0; v < 8; ++v) bfr[nt].u[v] = blds[apair(v, half) * 128 + n];
    }
#pragma unroll
    for (int mt = 0; mt < 2; ++mt)
#pragma unroll
      for (int nt = 0; nt < 4; ++nt)
        acc[mt][nt] = wmma_bf16(af[mt], bfr[nt], acc[mt][nt]);
    __syncthreads();
  }

  // Epilogue (C layout: lane=N, VGPR r -> M=r, +8 for upper half).
  // which/head/d/bias depend only on n -> hoist out of the row loop.
#pragma unroll
  for (int nt = 0; nt < 4; ++nt) {
    int n = bn * 128 + wc * 64 + nt * 16 + m16;
    int which = n / C_DIM, rem = n % C_DIM;     // which is scalar (768 % 128 == 0)
    int head = rem >> 6, d = rem & 63;
    float bv = bias[n];
#pragma unroll
    for (int mt = 0; mt < 2; ++mt) {
      int mbase = bm * 128 + wr * 32 + mt * 16 + 8 * half;
#pragma unroll
      for (int r = 0; r < 8; ++r) {
        int m = mbase + r;
        if (m >= M_TOK) continue;
        int b = m / TOK, t = m - b * TOK;
        int bh = b * NHEAD + head;
        uint16_t hv = f2bf(acc[mt][nt][r] + bv);
        if (which == 0)      qb [((size_t)bh * TOK + t) * HD + d] = hv;
        else if (which == 1) kTb[((size_t)bh * HD + d) * TOK + t] = hv;
        else                 vb [((size_t)bh * TOK + t) * HD + d] = hv;
      }
    }
  }
}

// ===========================================================================
// Kernel 2: attention per head-batch (grid = 2400).  256 threads = 8 waves.
// scores = 0.125*(q@kT) + rel_h + rel_w, softmax, out = P@V.
// ===========================================================================
__global__ __launch_bounds__(256) void fv_attention(
    const uint16_t* __restrict__ qb, const uint16_t* __restrict__ kTb,
    const uint16_t* __restrict__ vb,
    const float* __restrict__ relph, const float* __restrict__ relpw,
    uint16_t* __restrict__ ctx) {
  __shared__ uint32_t qlds[TOKP * 32];      // [row t][kpair]      28.0 KB
  __shared__ uint32_t klds[32 * TOKP];      // [cpair][col j]      28.0 KB
  __shared__ uint32_t vlds[112 * 64];       // [jpair][d]          28.0 KB
  __shared__ float    relh[TOK * 14];       //                     10.7 KB
  __shared__ float    relw[TOK * 14];       //                     10.7 KB
  __shared__ uint32_t plds[8 * 16 * 16];    // per-wave P chunk     8.0 KB

  const int tid = threadIdx.x, lane = tid & 31, wave = tid >> 5;
  const int bh = blockIdx.x;
  const int m16 = lane & 15, half = lane >> 4;

  // ---- stage Q (async DMA copy) / KT / V (pair-packing) into LDS ----
  {
    // Q rows are already packed bf16 pairs -> pure copy: async-to-LDS b128.
    const uint32_t* qg = (const uint32_t*)qb + (size_t)bh * TOK * 32;
    for (int idx = tid; idx < TOK * 8; idx += 256) {       // 1568 x 16B segs
      int off = idx * 4;
      async_g2l_b128(&qlds[off], qg + off);
    }
    for (int idx = tid; idx < 28 * 32; idx += 256)         // zero pad rows
      qlds[TOK * 32 + idx] = 0u;

    const uint16_t* kg = kTb + (size_t)bh * HD * TOK;
    for (int idx = tid; idx < 32 * TOKP; idx += 256) {
      int p = idx / TOKP, j = idx % TOKP;
      uint32_t w = 0u;
      if (j < TOK)
        w = (uint32_t)kg[(2 * p) * TOK + j] |
            ((uint32_t)kg[(2 * p + 1) * TOK + j] << 16);
      klds[idx] = w;
    }
    const uint16_t* vg = vb + (size_t)bh * TOK * HD;
    for (int idx = tid; idx < 112 * 64; idx += 256) {
      int jp = idx >> 6, d = idx & 63;
      int j0 = 2 * jp, j1 = 2 * jp + 1;
      uint32_t lo = (j0 < TOK) ? vg[j0 * HD + d] : 0u;
      uint32_t hi = (j1 < TOK) ? vg[j1 * HD + d] : 0u;
      vlds[idx] = lo | (hi << 16);
    }
  }
  wait_async0();        // this wave's async LDS writes done before signaling
  __syncthreads();

  // ---- decomposed rel-pos rows: rel_h[i][kh], rel_w[i][kw] (fp32) ----
  for (int idx = tid; idx < TOK * 28; idx += 256) {
    int i = idx / 28, rr = idx % 28;
    bool isH = rr < 14;
    int kk = isH ? rr : rr - 14;
    int h = i / 14, w = i % 14;
    int coord = (isH ? h : w) - kk + 13;              // in [0, 26]
    const float* rp = (isH ? relph : relpw) + coord * HD;
    const uint32_t* qrow = qlds + i * 32;
    float acc = 0.f;
#pragma unroll
    for (int pi = 0; pi < 32; ++pi) {
      uint32_t u = qrow[pi];
      acc += bf2f(u & 0xffffu) * rp[2 * pi] + bf2f(u >> 16) * rp[2 * pi + 1];
    }
    if (isH) relh[i * 14 + kk] = acc; else relw[i * 14 + kk] = acc;
  }
  __syncthreads();

  const int b = bh / NHEAD, head = bh % NHEAD;

  // ---- each wave owns 16-row strips ----
  for (int rt = wave; rt < 14; rt += 8) {
    // Q fragments for this strip (K=64 -> 2 chunks of 32)
    FragBF qa[2];
    {
      int qrow = rt * 16 + m16;
#pragma unroll
      for (int ch = 0; ch < 2; ++ch)
#pragma unroll
        for (int v = 0; v < 8; ++v)
          qa[ch].u[v] = qlds[qrow * 32 + ch * 16 + apair(v, half)];
    }
    // scores: 14 column tiles
    v8f S[14];
#pragma unroll
    for (int ct = 0; ct < 14; ++ct) {
      v8f acc = {};
      int jg = ct * 16 + m16;
#pragma unroll
      for (int ch = 0; ch < 2; ++ch) {
        FragBF kf;
#pragma unroll
        for (int v = 0; v < 8; ++v)
          kf.u[v] = klds[(ch * 16 + apair(v, half)) * TOKP + jg];
        acc = wmma_bf16(qa[ch], kf, acc);
      }
      S[ct] = acc;
    }
    // scale + bias + mask (kh/kw depend only on the lane's column -> hoist)
#pragma unroll
    for (int ct = 0; ct < 14; ++ct) {
      int jg = ct * 16 + m16;
      int kh = jg / 14, kw = jg % 14;
      bool jok = jg < TOK;
#pragma unroll
      for (int r = 0; r < 8; ++r) {
        int ig = rt * 16 + r + 8 * half;
        float s = -1e30f;
        if (jok && ig < TOK)
          s = S[ct][r] * 0.125f + relh[ig * 14 + kh] + relw[ig * 14 + kw];
        S[ct][r] = s;
      }
    }
    // softmax per row (rows live across the 16 lanes of each half-wave)
    float rmax[8], rsum[8];
#pragma unroll
    for (int r = 0; r < 8; ++r) {
      float mx = -1e30f;
#pragma unroll
      for (int ct = 0; ct < 14; ++ct) mx = fmaxf(mx, S[ct][r]);
#pragma unroll
      for (int msk = 1; msk < 16; msk <<= 1) mx = fmaxf(mx, __shfl_xor(mx, msk, 16));
      rmax[r] = mx;
    }
#pragma unroll
    for (int r = 0; r < 8; ++r) rsum[r] = 0.f;
#pragma unroll
    for (int ct = 0; ct < 14; ++ct)
#pragma unroll
      for (int r = 0; r < 8; ++r) {
        float e = __expf(S[ct][r] - rmax[r]);
        S[ct][r] = e;
        rsum[r] += e;
      }
#pragma unroll
    for (int r = 0; r < 8; ++r) {
      float sm = rsum[r];
#pragma unroll
      for (int msk = 1; msk < 16; msk <<= 1) sm += __shfl_xor(sm, msk, 16);
      rsum[r] = 1.f / sm;
    }

    // P @ V : K = 224 -> 7 chunks of 32, round-trip P through per-wave LDS
    v8f O[4];
    { v8f z = {}; O[0] = z; O[1] = z; O[2] = z; O[3] = z; }
    uint16_t* myp16 = (uint16_t*)(plds + wave * 16 * 16);
    const uint32_t* myp32 = plds + wave * 16 * 16;
#pragma unroll
    for (int kk = 0; kk < 7; ++kk) {
#pragma unroll
      for (int bb = 0; bb < 2; ++bb) {
        int ct = 2 * kk + bb;
#pragma unroll
        for (int r = 0; r < 8; ++r)
          myp16[(r + 8 * half) * 32 + bb * 16 + m16] = f2bf(S[ct][r] * rsum[r]);
      }
      asm volatile("s_wait_dscnt 0" ::: "memory");   // stores -> loads (same wave)
      FragBF pa;
#pragma unroll
      for (int v = 0; v < 8; ++v) pa.u[v] = myp32[m16 * 16 + apair(v, half)];
#pragma unroll
      for (int dt = 0; dt < 4; ++dt) {
        FragBF vf;
        int dg = dt * 16 + m16;
#pragma unroll
        for (int v = 0; v < 8; ++v)
          vf.u[v] = vlds[(kk * 16 + apair(v, half)) * 64 + dg];
        O[dt] = wmma_bf16(pa, vf, O[dt]);
      }
      asm volatile("s_wait_dscnt 0" ::: "memory");   // loads done before rewrite
    }

    // store context rows: ctx[b*196 + t][head*64 + d] bf16
#pragma unroll
    for (int dt = 0; dt < 4; ++dt)
#pragma unroll
      for (int r = 0; r < 8; ++r) {
        int t = rt * 16 + r + 8 * half;
        if (t >= TOK) continue;
        size_t off = ((size_t)(b * TOK + t)) * C_DIM + head * HD + dt * 16 + m16;
        ctx[off] = f2bf(O[dt][r]);
      }
  }
}

// ===========================================================================
// Kernel 3: output projection.  ctx(39200x768) bf16 @ Wproj(768x768) + bias
// -> fp32 out.  Same 128x128x32 WMMA tiling; A tile staged via async DMA.
// ===========================================================================
__global__ __launch_bounds__(256) void fv_proj_gemm(
    const uint16_t* __restrict__ ctx, const float* __restrict__ W,
    const float* __restrict__ bias, float* __restrict__ out) {
  __shared__ uint32_t alds[128 * 16];
  __shared__ uint32_t blds[16 * 128];

  const int tid = threadIdx.x, lane = tid & 31, wave = tid >> 5;
  const int bm = blockIdx.x, bn = blockIdx.y;
  const int wr = wave >> 1, wc = wave & 1;
  const int m16 = lane & 15, half = lane >> 4;

  v8f zero = {};
  v8f acc[2][4];
#pragma unroll
  for (int mt = 0; mt < 2; ++mt)
#pragma unroll
    for (int nt = 0; nt < 4; ++nt) acc[mt][nt] = zero;

  const uint32_t* ctx32 = (const uint32_t*)ctx;
  for (int k0 = 0; k0 < C_DIM; k0 += 32) {
    // A tile is already bf16 pairs in global -> async-to-LDS b128 copy.
#pragma unroll
    for (int idx = tid; idx < 512; idx += 256) {           // 128 rows x 4 segs
      int row = idx >> 2, seg = idx & 3;
      int m = bm * 128 + row;
      uint32_t* lp = &alds[row * 16 + seg * 4];
      if (m < M_TOK) {
        async_g2l_b128(lp, ctx32 + (size_t)m * (C_DIM / 2) + (k0 >> 1) + seg * 4);
      } else {
        lp[0] = 0u; lp[1] = 0u; lp[2] = 0u; lp[3] = 0u;
      }
    }
#pragma unroll
    for (int idx = tid; idx < 16 * 128; idx += 256) {
      int p = idx >> 7, n = idx & 127;
      int k = k0 + 2 * p, ng = bn * 128 + n;
      blds[idx] = pack2(W[(size_t)k * C_DIM + ng], W[(size_t)(k + 1) * C_DIM + ng]);
    }
    wait_async0();
    __syncthreads();

    FragBF af[2], bfr[4];
#pragma unroll
    for (int mt = 0; mt < 2; ++mt) {
      int row = wr * 32 + mt * 16 + m16;
#pragma unroll
      for (int v = 0; v < 8; ++v) af[mt].u[v] = alds[row * 16 + apair(v, half)];
    }
#pragma unroll
    for (int nt = 0; nt < 4; ++nt) {
      int n = wc * 64 + nt * 16 + m16;
#pragma unroll
      for (int v = 0; v < 8; ++v) bfr[nt].u[v] = blds[apair(v, half) * 128 + n];
    }
#pragma unroll
    for (int mt = 0; mt < 2; ++mt)
#pragma unroll
      for (int nt = 0; nt < 4; ++nt)
        acc[mt][nt] = wmma_bf16(af[mt], bfr[nt], acc[mt][nt]);
    __syncthreads();
  }

#pragma unroll
  for (int nt = 0; nt < 4; ++nt) {
    int n = bn * 128 + wc * 64 + nt * 16 + m16;
    float bv = bias[n];
#pragma unroll
    for (int mt = 0; mt < 2; ++mt) {
      int mbase = bm * 128 + wr * 32 + mt * 16 + 8 * half;
#pragma unroll
      for (int r = 0; r < 8; ++r) {
        int m = mbase + r;
        if (m >= M_TOK) continue;
        out[(size_t)m * C_DIM + n] = acc[mt][nt][r] + bv;
      }
    }
  }
}

// ===========================================================================
extern "C" void kernel_launch(void* const* d_in, const int* in_sizes, int n_in,
                              void* d_out, int out_size, void* d_ws, size_t ws_size,
                              hipStream_t stream) {
  const float* x       = (const float*)d_in[0];   // (200,14,14,768)
  const float* w_qkv   = (const float*)d_in[1];   // (768,2304)
  const float* b_qkv   = (const float*)d_in[2];   // (2304,)
  const float* w_proj  = (const float*)d_in[3];   // (768,768)
  const float* b_proj  = (const float*)d_in[4];   // (768,)
  const float* rel_h   = (const float*)d_in[5];   // (27,64)
  const float* rel_w   = (const float*)d_in[6];   // (27,64)
  float* out = (float*)d_out;

  // workspace carve-up (all bf16/uint16; ~241 MB total)
  const size_t nQKV = (size_t)2400 * TOK * HD;    // 30,105,600 elems each
  uint16_t* qb  = (uint16_t*)d_ws;
  uint16_t* kTb = qb  + nQKV;
  uint16_t* vb  = kTb + nQKV;
  uint16_t* ctx = vb  + nQKV;                     // 39200*768 elems

  dim3 blk(256);
  dim3 gQKV((M_TOK + 127) / 128, NQKV / 128);     // 307 x 18
  fv_qkv_gemm<<<gQKV, blk, 0, stream>>>(x, w_qkv, b_qkv, qb, kTb, vb);

  fv_attention<<<dim3(2400), blk, 0, stream>>>(qb, kTb, vb, rel_h, rel_w, ctx);

  dim3 gP((M_TOK + 127) / 128, C_DIM / 128);      // 307 x 6
  fv_proj_gemm<<<gP, blk, 0, stream>>>(ctx, w_proj, b_proj, out);
}
